// SpikeFP32LayerNorm_43860206027346
// MI455X (gfx1250) — matrix operand order repaired and checked by reading, not compiled, not run on hardware
//
#include <hip/hip_runtime.h>

// SpikeFP32LayerNorm for MI455X (gfx1250, wave32).
// Memory-bound streaming kernel (~1 GB @ 23.3 TB/s ~= 46 us floor).
// Bit-pulse decode is done with V_WMMA_F32_16X16X32_BF16: K=32 matches the 32
// IEEE-754 bit lanes; weights are exact powers of two in BF16, and hi/lo split
// keeps every accumulation exact in the f32 accumulator.

typedef __attribute__((ext_vector_type(4)))  float  v4f;
typedef __attribute__((ext_vector_type(8)))  float  v8f;
typedef __attribute__((ext_vector_type(16))) __bf16 v16bf;

union BFPack { unsigned u[8]; v16bf v; };

// Pack two exact {0.0f,1.0f} floats into a dword of two bf16 (truncation is exact).
__device__ __forceinline__ unsigned pack_bf16_pair(float lo, float hi) {
  unsigned a = __builtin_bit_cast(unsigned, lo);
  unsigned b = __builtin_bit_cast(unsigned, hi);
  return (a >> 16) | (b & 0xFFFF0000u);
}

// bf16 bit pattern of weight A[M][k]:
//   M==0: 2^(7-k)  for k<8   (hi byte of u)
//   M==1: 2^(31-k) for k>=8  (low 24 bits of u)
__device__ __forceinline__ unsigned weight_bits(int M, int k) {
  int e;
  if (M == 0)      { if (k >= 8) return 0u; e = 7  - k; }
  else if (M == 1) { if (k <  8) return 0u; e = 31 - k; }
  else return 0u;
  return (unsigned)(127 + e) << 7;   // bf16: exp field only, mantissa 0
}

// Build constant A (16x32 bf16) in contiguous k order (k = 16h + s), not the ISA table order:
// lanes 0-15: M=lane, VGPR j holds K pair (2j, 2j+1) (K 0..15);
// lanes 16-31: same M, K shifted by +16 (i.e. K 16..31).
__device__ __forceinline__ v16bf make_A(int lane) {
  BFPack A;
  const int M = lane & 15;
  const int h = lane >> 4;
#pragma unroll
  for (int j = 0; j < 8; ++j) {
    int k0 = 2 * j + (h ? 16 : 0);
    A.u[j] = weight_bits(M, k0) | (weight_bits(M, k0 + 1) << 16);
  }
  return A.v;
}

// Decode 16 consecutive values (columns of B). Lane l supplies B for column
// (l&15): bits [ (l>>4)*16, +16 ) of that value, in the same contiguous k order as A
// (lanes 0-15 hold K=0..15, lanes 16-31 hold K=16..31, two K per VGPR).
// Result valid in lanes 0..15: u for value (base + lane).
__device__ __forceinline__ unsigned decode16(const float* __restrict__ src, v16bf A) {
  BFPack B;
  const v4f* p = (const v4f*)src;
#pragma unroll
  for (int q = 0; q < 4; ++q) {
    v4f f = __builtin_nontemporal_load(p + q);
    B.u[2 * q + 0] = pack_bf16_pair(f.x, f.y);
    B.u[2 * q + 1] = pack_bf16_pair(f.z, f.w);
  }
  v8f c = {};
  c = __builtin_amdgcn_wmma_f32_16x16x32_bf16(
      /*neg_a=*/false, A, /*neg_b=*/false, B.v,
      /*c_mod=*/(short)0, c, /*reuse_a=*/false, /*reuse_b=*/false);
  // lanes 0..15: c[0] = hi byte (exact int < 2^8), c[1] = low 24 bits (< 2^24)
  return ((unsigned)c[0] << 24) | (unsigned)c[1];
}

__global__ __launch_bounds__(256)
void spike_fp32_layernorm_kernel(const float* __restrict__ x,
                                 float* __restrict__ out) {
  constexpr int N = 1024;
  const int row  = blockIdx.x;
  const int tid  = threadIdx.x;
  const int lane = tid & 31;
  const int wave = tid >> 5;
  const int n    = lane & 15;
  const int h    = lane >> 4;

  const v16bf A = make_A(lane);
  const size_t rowBase = (size_t)row * N * 32;

  __shared__ float smem[8];

  // ---- decode: each thread ends with 4 values (value index = it*256 + tid) ----
  float vv[4];
#pragma unroll
  for (int it = 0; it < 4; ++it) {
    const int vb = it * 256 + wave * 32;
    const float* p1 = x + rowBase + (size_t)(vb + n) * 32 + (size_t)h * 16;
    const float* p2 = p1 + 16 * 32;               // values vb+16 .. vb+31
    unsigned u1 = decode16(p1, A);                // valid lanes 0..15
    unsigned u2 = decode16(p2, A);                // valid lanes 0..15
    unsigned u2s = (unsigned)__shfl((int)u2, n);  // lane l>=16 pulls from l-16
    unsigned u = (lane < 16) ? u1 : u2s;          // lane l holds value vb+l
    vv[it] = __builtin_bit_cast(float, u);
  }

  // ---- mean (wave32 butterfly + LDS across 8 waves) ----
  float s = vv[0] + vv[1] + vv[2] + vv[3];
#pragma unroll
  for (int off = 16; off > 0; off >>= 1) s += __shfl_xor(s, off);
  if (lane == 0) smem[wave] = s;
  __syncthreads();
  float S = 0.f;
#pragma unroll
  for (int w = 0; w < 8; ++w) S += smem[w];
  const float mean = S * (1.0f / 1024.0f);

  // ---- biased variance of centered values ----
  float d = 0.f;
#pragma unroll
  for (int i = 0; i < 4; ++i) { float t = vv[i] - mean; d += t * t; }
#pragma unroll
  for (int off = 16; off > 0; off >>= 1) d += __shfl_xor(d, off);
  __syncthreads();                       // smem reuse
  if (lane == 0) smem[wave] = d;
  __syncthreads();
  float S2 = 0.f;
#pragma unroll
  for (int w = 0; w < 8; ++w) S2 += smem[w];
  const float rstd = rsqrtf(S2 * (1.0f / 1024.0f) + 1e-6f);

  // ---- normalize + re-encode to bit pulses (8 NT b128 stores per value) ----
#pragma unroll
  for (int it = 0; it < 4; ++it) {
    const float y = (vv[it] - mean) * rstd;
    const unsigned u = __builtin_bit_cast(unsigned, y);
    v4f* q4 = (v4f*)(out + rowBase + (size_t)(it * 256 + tid) * 32);
#pragma unroll
    for (int g = 0; g < 8; ++g) {
      v4f f;
#pragma unroll
      for (int e = 0; e < 4; ++e) {
        const int k = g * 4 + e;                       // MSB first
        f[e] = (float)((u >> (31 - k)) & 1u);
      }
      __builtin_nontemporal_store(f, q4 + g);
    }
  }
}

extern "C" void kernel_launch(void* const* d_in, const int* in_sizes, int n_in,
                              void* d_out, int out_size, void* d_ws, size_t ws_size,
                              hipStream_t stream) {
  (void)n_in; (void)out_size; (void)d_ws; (void)ws_size;
  const float* x = (const float*)d_in[0];
  float* out = (float*)d_out;
  const int rows = in_sizes[0] >> 15;   // / (1024*32) == 4096
  spike_fp32_layernorm_kernel<<<rows, 256, 0, stream>>>(x, out);
}
